// Net_58033598104036
// MI455X (gfx1250) — matrix operand order repaired and checked by reading, not compiled, not run on hardware
//
#include <hip/hip_runtime.h>
#include <math.h>

typedef _Float16 v16h __attribute__((ext_vector_type(16)));
typedef _Float16 v8h  __attribute__((ext_vector_type(8)));
typedef float    v8f  __attribute__((ext_vector_type(8)));

typedef unsigned int u32x4 __attribute__((ext_vector_type(4)));
typedef int          i32x8 __attribute__((ext_vector_type(8)));
typedef int          i32x4 __attribute__((ext_vector_type(4)));

#define EDGE_WAVES 2

// ---------------------------------------------------------------------------
// Native CDNA5 non-returning float atomic max (STOREcnt-tracked).
// ---------------------------------------------------------------------------
__device__ __forceinline__ void atomicMaxF32(float* addr, float val) {
  asm volatile("global_atomic_max_num_f32 %0, %1, off"
               :: "v"(addr), "v"(val) : "memory");
}

// ---------------------------------------------------------------------------
// TDM: 1-D tile load Global -> LDS via TENSOR_LOAD_TO_LDS (D# per ISA 8.3/8.4).
// nbytes must be a multiple of 8 and nbytes/8 <= 65535 (16-bit tile_dim0).
// lds_addr is the wave-relative LDS byte address (flat low 32 bits).
// 6-arg builtin variant (clang-23 / therock headers).
// ---------------------------------------------------------------------------
__device__ __forceinline__ void tdm_load_1d(unsigned lds_addr, const void* gptr,
                                            unsigned nbytes) {
  unsigned n = nbytes >> 3;                      // 8-byte elements
  unsigned long long ga = (unsigned long long)(uintptr_t)gptr;
  u32x4 g0;
  g0[0] = 1u;                                    // count=1, user mode
  g0[1] = lds_addr;                              // D#.lds_addr
  g0[2] = (unsigned)ga;                          // global_addr[31:0]
  g0[3] = (unsigned)((ga >> 32) & 0x01FFFFFFu)   // global_addr[56:32]
        | 0x80000000u;                           // type=2 ("image")
  i32x8 g1;
  g1[0] = (int)(3u << 16);                       // data_size=8B; no multicast
  g1[1] = (int)((n & 0xFFFFu) << 16);            // tensor_dim0[15:0]
  g1[2] = (int)((n >> 16) | (1u << 16));         // tensor_dim0[31:16], tensor_dim1=1
  g1[3] = (int)((n & 0xFFFFu) << 16);            // tile_dim0 = n
  g1[4] = 0;                                     // tile_dim1=0, tile_dim2=0 (unused)
  g1[5] = (int)n;                                // tensor_dim0_stride[31:0]
  g1[6] = 0;
  g1[7] = 0;
  i32x4 z4 = {0, 0, 0, 0};
  i32x8 z8 = {0, 0, 0, 0, 0, 0, 0, 0};
  __builtin_amdgcn_tensor_load_to_lds(g0, g1, z4, z4, z8, 0);
}

// ---------------------------------------------------------------------------
// Generic fills / fixups
// ---------------------------------------------------------------------------
__global__ void fill_kernel(float* __restrict__ p, float v, long long n) {
  long long i = (long long)blockIdx.x * blockDim.x + threadIdx.x;
  if (i < n) p[i] = v;
}

__global__ void finite_or_zero_kernel(float* __restrict__ p, long long n) {
  long long i = (long long)blockIdx.x * blockDim.x + threadIdx.x;
  if (i < n) {
    float v = p[i];
    p[i] = __builtin_isfinite(v) ? v : 0.0f;
  }
}

// ---------------------------------------------------------------------------
// Per-node factorization of EdgeConv layer 1:
//   base_i = pos_i @ (W1[0:3] - W1[3:6]) + x_i @ W1[6:] + b1
//   s_j    = pos_j @ W1[3:6]
// ---------------------------------------------------------------------------
template<int C, int D1>
__global__ __launch_bounds__(256)
void node_pre_kernel(const float* __restrict__ pos, const float* __restrict__ x,
                     const float* __restrict__ W1, const float* __restrict__ b1,
                     float* __restrict__ base, float* __restrict__ sbuf, int N) {
  int gid = blockIdx.x * blockDim.x + threadIdx.x;
  if (gid >= N * D1) return;
  int i = gid / D1, c = gid % D1;
  float p0 = pos[i * 3 + 0], p1 = pos[i * 3 + 1], p2 = pos[i * 3 + 2];
  float sv  = p0 * W1[3 * D1 + c] + p1 * W1[4 * D1 + c] + p2 * W1[5 * D1 + c];
  float acc = b1[c] + p0 * W1[0 * D1 + c] + p1 * W1[1 * D1 + c] + p2 * W1[2 * D1 + c] - sv;
#pragma unroll
  for (int k = 0; k < C; ++k) acc += x[i * C + k] * W1[(6 + k) * D1 + c];
  base[gid] = acc;
  sbuf[gid] = sv;
}

// ---------------------------------------------------------------------------
// Fold BN into next linear and pre-swizzle weights into the exact WMMA-B
// per-lane layout: for K-chunk kc, N-tile nt, lane L, element i:
//   k = kc*32 + (L/16)*8 + (i%8) + (i/8)*16,  n = nt*16 + (L%16)
// so each lane's v16h B fragment is a contiguous 32-byte LDS read.
// ---------------------------------------------------------------------------
template<int D1, int D2, int D3>
__global__ __launch_bounds__(256)
void prep_layer_kernel(const float* __restrict__ W2, const float* __restrict__ b2,
                       const float* __restrict__ g1, const float* __restrict__ bt1,
                       const float* __restrict__ mn1, const float* __restrict__ vr1,
                       const float* __restrict__ g2, const float* __restrict__ bt2,
                       const float* __restrict__ mn2, const float* __restrict__ vr2,
                       const float* __restrict__ W3, const float* __restrict__ b3,
                       const float* __restrict__ g3, const float* __restrict__ bt3,
                       const float* __restrict__ mn3, const float* __restrict__ vr3,
                       _Float16* __restrict__ wB2, float* __restrict__ b2f,
                       _Float16* __restrict__ wB3, float* __restrict__ b3f,
                       float* __restrict__ sc3, float* __restrict__ sh3) {
  __shared__ float s1[D1], t1[D1], s2[D2], t2[D2];
  int tid = threadIdx.x;
  if (tid < D1) { float sc = g1[tid] * rsqrtf(vr1[tid] + 1e-5f); s1[tid] = sc; t1[tid] = bt1[tid] - mn1[tid] * sc; }
  if (tid < D2) { float sc = g2[tid] * rsqrtf(vr2[tid] + 1e-5f); s2[tid] = sc; t2[tid] = bt2[tid] - mn2[tid] * sc; }
  if (tid < D3) { float sc = g3[tid] * rsqrtf(vr3[tid] + 1e-5f); sc3[tid] = sc; sh3[tid] = bt3[tid] - mn3[tid] * sc; }
  __syncthreads();

  for (int t = tid; t < D1 * D2; t += 256) {
    int i = t & 15, lane = (t >> 4) & 31, blk = t >> 9;
    int nt = blk % (D2 / 16), kc = blk / (D2 / 16);
    int k = kc * 32 + (lane >> 4) * 8 + (i & 7) + (i >> 3) * 16;
    int n = nt * 16 + (lane & 15);
    wB2[t] = (_Float16)(s1[k] * W2[k * D2 + n]);
  }
  for (int n = tid; n < D2; n += 256) {
    float a = b2[n];
    for (int k = 0; k < D1; ++k) a += t1[k] * W2[k * D2 + n];
    b2f[n] = a;
  }
  for (int t = tid; t < D2 * D3; t += 256) {
    int i = t & 15, lane = (t >> 4) & 31, blk = t >> 9;
    int nt = blk % (D3 / 16), kc = blk / (D3 / 16);
    int k = kc * 32 + (lane >> 4) * 8 + (i & 7) + (i >> 3) * 16;
    int n = nt * 16 + (lane & 15);
    wB3[t] = (_Float16)(s2[k] * W3[k * D3 + n]);
  }
  for (int n = tid; n < D3; n += 256) {
    float a = b3[n];
    for (int k = 0; k < D2; ++k) a += t2[k] * W3[k * D3 + n];
    b3f[n] = a;
  }
}

// ---------------------------------------------------------------------------
// Hot kernel (persistent tiles): TDM-stage folded weights to LDS once per
// block, then per 16-edge tile run two WMMA GEMM chains + BN3 affine +
// native atomic-max scatter to out[dst].
// A fragment (16-bit A 16x32): lane L holds row M=L%16,
//   K = (L/16)*8 + {0..7} and (L/16)*8 + 16 + {0..7}  -> two ds_load_b128.
// C/D layout: VGPR r -> M = r + (L/16)*8, N = nt*16 + L%16.
// ---------------------------------------------------------------------------
template<int D1, int D2, int D3>
__global__ __launch_bounds__(EDGE_WAVES * 32)
void edge_mlp_kernel(const int* __restrict__ ei, int E,
                     const float* __restrict__ base, const float* __restrict__ sbuf,
                     const _Float16* __restrict__ wB2, const float* __restrict__ b2f,
                     const _Float16* __restrict__ wB3, const float* __restrict__ b3f,
                     const float* __restrict__ sc3, const float* __restrict__ sh3,
                     float* __restrict__ out) {
  __shared__ __align__(32) _Float16 sW2[D1 * D2];
  __shared__ __align__(32) _Float16 sW3[D2 * D3];
  __shared__ __align__(32) _Float16 sA1[EDGE_WAVES][16 * D1];
  __shared__ __align__(32) _Float16 sA2[EDGE_WAVES][16 * D2];
  __shared__ int sSrc[EDGE_WAVES][16];
  __shared__ int sDst[EDGE_WAVES][16];

  const int lane = threadIdx.x & 31;
  const int w    = threadIdx.x >> 5;

  // Stage both folded weight matrices into LDS with the Tensor Data Mover.
  if (w == 0) {
    tdm_load_1d((unsigned)(uintptr_t)(void*)sW2, wB2, (unsigned)(D1 * D2 * 2));
    tdm_load_1d((unsigned)(uintptr_t)(void*)sW3, wB3, (unsigned)(D2 * D3 * 2));
    __builtin_amdgcn_s_wait_tensorcnt(0);
  }
  __syncthreads();

  const int tiles  = (E + 15) >> 4;
  const int nWaves = (int)gridDim.x * EDGE_WAVES;
  const int iters  = (tiles + nWaves - 1) / nWaves;
  const int myWave = (int)blockIdx.x * EDGE_WAVES + w;

  const int row = lane & 15;
  const int kb  = (lane >> 4) * 8;
  const int mb  = (lane >> 4) * 8;

  for (int it = 0; it < iters; ++it) {
    int tile = myWave + it * nWaves;
    if (tile >= tiles) tile = tiles - 1;   // duplicated tile harmless under max-agg
    const int e0 = tile << 4;

    __syncthreads();                        // staging buffers free to overwrite
    if (lane < 16) {
      int e = e0 + lane; if (e >= E) e = E - 1;
      sSrc[w][lane] = ei[e];
      sDst[w][lane] = ei[E + e];
    }
    __syncthreads();

    // Stage h1 = relu(base[dst] + s[src]) as f16, row-major [16][D1].
    for (int t = 0; t < (16 * D1) / 32; ++t) {
      int idx = lane + t * 32;
      int m = idx / D1, c = idx % D1;       // consecutive lanes -> consecutive c
      float v = base[(size_t)sDst[w][m] * D1 + c] + sbuf[(size_t)sSrc[w][m] * D1 + c];
      sA1[w][idx] = (_Float16)fmaxf(v, 0.0f);
    }
    __syncthreads();

    // GEMM2: [16 x D1] @ [D1 x D2], relu, stage to sA2 (f16).
#pragma unroll
    for (int nt = 0; nt < D2 / 16; ++nt) {
      v8f acc = {};
#pragma unroll
      for (int kc = 0; kc < D1 / 32; ++kc) {
        const _Float16* ap = &sA1[w][row * D1 + kc * 32 + kb];
        v8h alo = *(const v8h*)(ap);
        v8h ahi = *(const v8h*)(ap + 16);
        v16h A;
#pragma unroll
        for (int i = 0; i < 8; ++i) { A[i] = alo[i]; A[i + 8] = ahi[i]; }
        const _Float16* bp = &sW2[((kc * (D2 / 16) + nt) * 32 + lane) * 16];
        v8h blo = *(const v8h*)(bp);
        v8h bhi = *(const v8h*)(bp + 8);
        v16h B;
#pragma unroll
        for (int i = 0; i < 8; ++i) { B[i] = blo[i]; B[i + 8] = bhi[i]; }
        acc = __builtin_amdgcn_wmma_f32_16x16x32_f16(false, A, false, B, (short)0, acc,
                                                     false, false);
      }
      int n = nt * 16 + row;
      float bb = b2f[n];
#pragma unroll
      for (int r = 0; r < 8; ++r) {
        float z = acc[r] + bb;
        sA2[w][(mb + r) * D2 + n] = (_Float16)fmaxf(z, 0.0f);
      }
    }
    __syncthreads();

    // GEMM3: [16 x D2] @ [D2 x D3], relu, BN3 affine, atomic scatter-max.
#pragma unroll
    for (int nt = 0; nt < D3 / 16; ++nt) {
      v8f acc = {};
#pragma unroll
      for (int kc = 0; kc < D2 / 32; ++kc) {
        const _Float16* ap = &sA2[w][row * D2 + kc * 32 + kb];
        v8h alo = *(const v8h*)(ap);
        v8h ahi = *(const v8h*)(ap + 16);
        v16h A;
#pragma unroll
        for (int i = 0; i < 8; ++i) { A[i] = alo[i]; A[i + 8] = ahi[i]; }
        const _Float16* bp = &sW3[((kc * (D3 / 16) + nt) * 32 + lane) * 16];
        v8h blo = *(const v8h*)(bp);
        v8h bhi = *(const v8h*)(bp + 8);
        v16h B;
#pragma unroll
        for (int i = 0; i < 8; ++i) { B[i] = blo[i]; B[i + 8] = bhi[i]; }
        acc = __builtin_amdgcn_wmma_f32_16x16x32_f16(false, A, false, B, (short)0, acc,
                                                     false, false);
      }
      int n = nt * 16 + row;
      float bb = b3f[n], sc = sc3[n], sh = sh3[n];
#pragma unroll
      for (int r = 0; r < 8; ++r) {
        float h = fmaxf(acc[r] + bb, 0.0f);
        float val = h * sc + sh;
        atomicMaxF32(&out[(size_t)sDst[w][mb + r] * D3 + n], val);
      }
    }
  }
}

// ---------------------------------------------------------------------------
// Global max-pool by graph id (atomic max into [G,128]).
// ---------------------------------------------------------------------------
__global__ void pool_kernel(const float* __restrict__ feat, const int* __restrict__ batch,
                            float* __restrict__ g, long long n) {
  long long gid = (long long)blockIdx.x * blockDim.x + threadIdx.x;
  if (gid >= n) return;
  int i = (int)(gid / 128), c = (int)(gid % 128);
  atomicMaxF32(&g[(size_t)batch[i] * 128 + c], feat[gid]);
}

// ---------------------------------------------------------------------------
// FC head: finite-fix pooled row, lin1(128->64)+relu, lin2(64->32)+relu,
// lin3(32->40), log_softmax.  One block per graph.
// ---------------------------------------------------------------------------
__global__ __launch_bounds__(128)
void head_kernel(const float* __restrict__ g,
                 const float* __restrict__ W1, const float* __restrict__ b1,
                 const float* __restrict__ W2, const float* __restrict__ b2,
                 const float* __restrict__ W3, const float* __restrict__ b3,
                 float* __restrict__ out) {
  __shared__ float gg[128], h1[64], h2[32], z[40], red[2];
  int t = threadIdx.x;
  float v = g[(size_t)blockIdx.x * 128 + t];
  gg[t] = __builtin_isfinite(v) ? v : 0.0f;
  __syncthreads();
  if (t < 64) {
    float a = b1[t];
#pragma unroll
    for (int k = 0; k < 128; ++k) a += gg[k] * W1[k * 64 + t];
    h1[t] = fmaxf(a, 0.0f);
  }
  __syncthreads();
  if (t < 32) {
    float a = b2[t];
#pragma unroll
    for (int k = 0; k < 64; ++k) a += h1[k] * W2[k * 32 + t];
    h2[t] = fmaxf(a, 0.0f);
  }
  __syncthreads();
  if (t < 40) {
    float a = b3[t];
#pragma unroll
    for (int k = 0; k < 32; ++k) a += h2[k] * W3[k * 40 + t];
    z[t] = a;
  }
  __syncthreads();
  if (t == 0) {
    float mx = -INFINITY;
    for (int i = 0; i < 40; ++i) mx = fmaxf(mx, z[i]);
    float sm = 0.0f;
    for (int i = 0; i < 40; ++i) sm += expf(z[i] - mx);
    red[0] = mx; red[1] = logf(sm);
  }
  __syncthreads();
  if (t < 40) out[blockIdx.x * 40 + t] = z[t] - red[0] - red[1];
}

// ---------------------------------------------------------------------------
// Host-side per-conv driver
// ---------------------------------------------------------------------------
template<int C, int D1, int D2, int D3>
static void run_conv(const float* xin, const float* pos, const int* ei, int N, int E,
                     const float* const* P,      // 18 arrays: 3 layers x (W,b,g,beta,mean,var)
                     float* base, float* sbuf, float* feat,
                     _Float16* wB2, float* b2f, _Float16* wB3, float* b3f,
                     float* sc3, float* sh3, hipStream_t stream) {
  prep_layer_kernel<D1, D2, D3><<<1, 256, 0, stream>>>(
      P[6 + 0], P[6 + 1],
      P[0 + 2], P[0 + 3], P[0 + 4], P[0 + 5],
      P[6 + 2], P[6 + 3], P[6 + 4], P[6 + 5],
      P[12 + 0], P[12 + 1],
      P[12 + 2], P[12 + 3], P[12 + 4], P[12 + 5],
      wB2, b2f, wB3, b3f, sc3, sh3);

  node_pre_kernel<C, D1><<<(N * D1 + 255) / 256, 256, 0, stream>>>(
      pos, xin, P[0], P[1], base, sbuf, N);

  long long nOut = (long long)N * D3;
  fill_kernel<<<(int)((nOut + 255) / 256), 256, 0, stream>>>(feat, -INFINITY, nOut);

  int tiles = (E + 15) / 16;
  int needBlocks = (tiles + EDGE_WAVES - 1) / EDGE_WAVES;
  int blocks = needBlocks < 2048 ? needBlocks : 2048;
  edge_mlp_kernel<D1, D2, D3><<<blocks, EDGE_WAVES * 32, 0, stream>>>(
      ei, E, base, sbuf, wB2, b2f, wB3, b3f, sc3, sh3, feat);

  finite_or_zero_kernel<<<(int)((nOut + 255) / 256), 256, 0, stream>>>(feat, nOut);
}

// ---------------------------------------------------------------------------
extern "C" void kernel_launch(void* const* d_in, const int* in_sizes, int n_in,
                              void* d_out, int out_size, void* d_ws, size_t ws_size,
                              hipStream_t stream) {
  (void)n_in; (void)ws_size;
  const float* x     = (const float*)d_in[0];
  const float* pos   = (const float*)d_in[1];
  const int*   ei    = (const int*)d_in[2];
  const int*   batch = (const int*)d_in[3];

  const int N = in_sizes[1] / 3;
  const int E = in_sizes[2] / 2;
  const int G = out_size / 40;

  const float* P1[18]; const float* P2[18]; const float* P3[18];
  for (int i = 0; i < 18; ++i) {
    P1[i] = (const float*)d_in[4 + i];
    P2[i] = (const float*)d_in[22 + i];
    P3[i] = (const float*)d_in[40 + i];
  }
  const float* l1W = (const float*)d_in[58]; const float* l1b = (const float*)d_in[59];
  const float* l2W = (const float*)d_in[60]; const float* l2b = (const float*)d_in[61];
  const float* l3W = (const float*)d_in[62]; const float* l3b = (const float*)d_in[63];

  // Workspace carve (256B-aligned regions)
  char* ws = (char*)d_ws;
  size_t off = 0;
  auto carve = [&](size_t bytes) -> void* {
    void* p = ws + off;
    off = (off + bytes + 255) & ~(size_t)255;
    return p;
  };
  float*    feat  = (float*)carve((size_t)N * 128 * sizeof(float));
  float*    base  = (float*)carve((size_t)N * 128 * sizeof(float));
  float*    sbuf  = (float*)carve((size_t)N * 128 * sizeof(float));
  float*    gpool = (float*)carve((size_t)G * 128 * sizeof(float));
  _Float16* wB2   = (_Float16*)carve(128 * 128 * sizeof(_Float16));
  float*    b2f   = (float*)carve(128 * sizeof(float));
  _Float16* wB3   = (_Float16*)carve(128 * 128 * sizeof(_Float16));
  float*    b3f   = (float*)carve(128 * sizeof(float));
  float*    sc3   = (float*)carve(128 * sizeof(float));
  float*    sh3   = (float*)carve(128 * sizeof(float));

  run_conv<5,  32, 32, 32 >(x,    pos, ei, N, E, P1, base, sbuf, feat, wB2, b2f, wB3, b3f, sc3, sh3, stream);
  run_conv<32, 64, 64, 64 >(feat, pos, ei, N, E, P2, base, sbuf, feat, wB2, b2f, wB3, b3f, sc3, sh3, stream);
  run_conv<64, 64, 128,128>(feat, pos, ei, N, E, P3, base, sbuf, feat, wB2, b2f, wB3, b3f, sc3, sh3, stream);

  long long nPool = (long long)G * 128;
  fill_kernel<<<(int)((nPool + 255) / 256), 256, 0, stream>>>(gpool, -INFINITY, nPool);
  long long nFeat = (long long)N * 128;
  pool_kernel<<<(int)((nFeat + 255) / 256), 256, 0, stream>>>(feat, batch, gpool, nFeat);

  head_kernel<<<G, 128, 0, stream>>>(gpool, l1W, l1b, l2W, l2b, l3W, l3b, (float*)d_out);
}
